// GCN_37847251812697
// MI455X (gfx1250) — matrix operand order repaired and checked by reading, not compiled, not run on hardware
//
#include <hip/hip_runtime.h>
#include <hip/hip_bf16.h>

typedef __attribute__((ext_vector_type(16))) __bf16 v16bf;
typedef __attribute__((ext_vector_type(8)))  float  v8f;

union FragBF {
    v16bf v;
    unsigned int u[8];
};

__device__ __forceinline__ unsigned short f2bf(float f) {
    unsigned int u = __float_as_uint(f);
    // round-to-nearest-even bf16 truncation
    unsigned int r = (u + 0x7FFFu + ((u >> 16) & 1u)) >> 16;
    return (unsigned short)r;
}

// ---------------- utility: zero a float region (grid-stride) ----------------
__global__ void zero_f32(float* __restrict__ p, long long n) {
    long long i = (long long)blockIdx.x * blockDim.x + threadIdx.x;
    long long stride = (long long)gridDim.x * blockDim.x;
    for (; i < n; i += stride) p[i] = 0.0f;
}

// ---------------- degree accumulation ----------------
__global__ void degree_kernel(const int* __restrict__ src, const int* __restrict__ dst,
                              float* __restrict__ deg_out, float* __restrict__ deg_in, int E) {
    int e = blockIdx.x * blockDim.x + threadIdx.x;
    if (e >= E) return;
    atomicAdd(deg_out + src[e], 1.0f);
    atomicAdd(deg_in  + dst[e], 1.0f);
}

// in-place: deg -> rsqrt(max(deg,1))
__global__ void norm_kernel(float* __restrict__ a, float* __restrict__ b, int N) {
    int i = blockIdx.x * blockDim.x + threadIdx.x;
    if (i >= N) return;
    a[i] = rsqrtf(fmaxf(a[i], 1.0f));
    b[i] = rsqrtf(fmaxf(b[i], 1.0f));
}

// ---------------- edge scatter, dim 16 (4 threads per edge, float4/lane) ----------------
__global__ void scatter16(const float* __restrict__ x, const float* __restrict__ nsrc,
                          const int* __restrict__ src, const int* __restrict__ dst,
                          float* __restrict__ agg, int E) {
    int tid = blockIdx.x * blockDim.x + threadIdx.x;
    int e = tid >> 2;
    if (e >= E) return;
    int f = (tid & 3) * 4;
    int s = src[e], d = dst[e];
    float ns = nsrc[s];
    float4 v = *(const float4*)(x + (size_t)s * 16 + f);
    float* out = agg + (size_t)d * 16 + f;
    atomicAdd(out + 0, v.x * ns);
    atomicAdd(out + 1, v.y * ns);
    atomicAdd(out + 2, v.z * ns);
    atomicAdd(out + 3, v.w * ns);
}

// ---------------- edge scatter, dim 128 (one wave32 per edge, float4/lane) ----------------
__global__ void scatter128(const float* __restrict__ x, const float* __restrict__ nsrc,
                           const int* __restrict__ src, const int* __restrict__ dst,
                           float* __restrict__ agg, int E) {
    int e = blockIdx.x * (blockDim.x >> 5) + (threadIdx.x >> 5);
    if (e >= E) return;
    int lane = threadIdx.x & 31;
    int s = src[e], d = dst[e];
    float ns = nsrc[s];
    float4 v = ((const float4*)(x + (size_t)s * 128))[lane];
    float* out = agg + (size_t)d * 128 + lane * 4;
    atomicAdd(out + 0, v.x * ns);
    atomicAdd(out + 1, v.y * ns);
    atomicAdd(out + 2, v.z * ns);
    atomicAdd(out + 3, v.w * ns);
}

// ---------------- WMMA GEMM: out[N,128] = relu?( (A * normd[row]) @ W + b ) ----------------
// KIN = logical K (16 or 128), KP = padded K (32 or 128), multiple of 32.
// Block: 128 threads = 4 waves. Each wave computes 16 rows x 128 cols.
template <int KIN, int KP, bool RELU>
__global__ __launch_bounds__(128) void gemm_norm_bias(
    const float* __restrict__ A,     // [N, KIN]
    const float* __restrict__ normd, // [N]
    const float* __restrict__ W,     // [KIN, 128]
    const float* __restrict__ bias,  // [128]
    float* __restrict__ out,         // [N, 128]
    int N)
{
    __shared__ __align__(16) unsigned short Wlds[128 * KP];     // transposed: [n][k]
    __shared__ __align__(16) unsigned short Alds[4 * 16 * KP];  // per-wave [16][k]

    const int tid  = threadIdx.x;
    const int wave = tid >> 5;
    const int lane = tid & 31;
    const int rowBase = blockIdx.x * 64;

    // Stage W transposed into LDS as bf16 (zero-pad K beyond KIN)
    for (int i = tid; i < 128 * KP; i += 128) {
        int n = i / KP, k = i % KP;
        float w = (k < KIN) ? W[(size_t)k * 128 + n] : 0.0f;
        Wlds[i] = f2bf(w);
    }

    // Each wave stages its own 16 rows of A, pre-scaled by norm_dst, bf16
    const int aBase = wave * 16 * KP;
    for (int i = lane; i < 16 * KP; i += 32) {
        int m = i / KP, k = i % KP;
        int row = rowBase + wave * 16 + m;
        float v = 0.0f;
        if (row < N && k < KIN) v = A[(size_t)row * KIN + k] * normd[row];
        Alds[aBase + i] = f2bf(v);
    }
    __syncthreads();

    v8f acc[8];
#pragma unroll
    for (int i = 0; i < 8; ++i)
#pragma unroll
        for (int r = 0; r < 8; ++r) acc[i][r] = 0.0f;

    // Fragment index math per ISA 16-bit A 16x32 / B 32x16 lane layouts
    const int mA    = lane & 15;
    const int koffA = (lane >> 4) << 3;   // 0 or 8
    const int koffB = (lane >> 4) << 3;   // 0 or 8
    const int nB    = lane & 15;

#pragma unroll
    for (int kk = 0; kk < KP / 32; ++kk) {
        FragBF a;
#pragma unroll
        for (int j = 0; j < 8; ++j) {
            int k = kk * 32 + koffA + ((j < 4) ? (2 * j) : (16 + 2 * (j - 4)));
            a.u[j] = *(const unsigned int*)&Alds[aBase + mA * KP + k];
        }
#pragma unroll
        for (int nt = 0; nt < 8; ++nt) {
            FragBF b;
#pragma unroll
            for (int j = 0; j < 8; ++j) {
                int k = kk * 32 + koffB + ((j < 4) ? (2 * j) : (16 + 2 * (j - 4)));
                b.u[j] = *(const unsigned int*)&Wlds[(nt * 16 + nB) * KP + k];
            }
            acc[nt] = __builtin_amdgcn_wmma_f32_16x16x32_bf16(
                false, a.v, false, b.v, (short)0, acc[nt], false, false);
        }
    }

    // D layout: VGPR r, lanes 0-15 -> M=r, lanes 16-31 -> M=8+r; N = lane&15
    const int rhi = (lane >> 4) * 8;
    const int wrow = rowBase + wave * 16 + rhi;      // first of 8 consecutive rows
    if (rowBase + 64 <= N) {
        // fast path: whole block in range -> unguarded stores with immediate offsets
#pragma unroll
        for (int nt = 0; nt < 8; ++nt) {
            int col = nt * 16 + (lane & 15);
            float bv = bias[col];
            float* p = out + (size_t)wrow * 128 + col;
#pragma unroll
            for (int r = 0; r < 8; ++r) {
                float v = acc[nt][r] + bv;
                if (RELU) v = fmaxf(v, 0.0f);
                p[(size_t)r * 128] = v;
            }
        }
    } else {
#pragma unroll
        for (int nt = 0; nt < 8; ++nt) {
            int col = nt * 16 + (lane & 15);
            float bv = bias[col];
#pragma unroll
            for (int r = 0; r < 8; ++r) {
                int row = wrow + r;
                if (row < N) {
                    float v = acc[nt][r] + bv;
                    if (RELU) v = fmaxf(v, 0.0f);
                    out[(size_t)row * 128 + col] = v;
                }
            }
        }
    }
}

// ---------------- layer 4: per-node dot with W4 (wave per node) ----------------
__global__ void dot128(const float* __restrict__ x, const float* __restrict__ w4,
                       float* __restrict__ t, int N) {
    int node = blockIdx.x * (blockDim.x >> 5) + (threadIdx.x >> 5);
    if (node >= N) return;
    int lane = threadIdx.x & 31;
    float4 xv = ((const float4*)(x + (size_t)node * 128))[lane];
    float4 wv = ((const float4*)w4)[lane];
    float p = xv.x * wv.x + xv.y * wv.y + xv.z * wv.z + xv.w * wv.w;
#pragma unroll
    for (int off = 16; off > 0; off >>= 1) p += __shfl_down(p, off, 32);
    if (lane == 0) t[node] = p;
}

// ---------------- layer 4: scalar edge scatter ----------------
__global__ void scatter1(const float* __restrict__ t, const float* __restrict__ nsrc,
                         const int* __restrict__ src, const int* __restrict__ dst,
                         float* __restrict__ s4, int E) {
    int e = blockIdx.x * blockDim.x + threadIdx.x;
    if (e >= E) return;
    int s = src[e];
    atomicAdd(s4 + dst[e], t[s] * nsrc[s]);
}

// ---------------- per-graph mean pooling ----------------
__global__ void pool_kernel(const float* __restrict__ s4, const float* __restrict__ ndst,
                            const int* __restrict__ graph_id, const float* __restrict__ b4,
                            float* __restrict__ gsum, float* __restrict__ gcnt, int N) {
    int i = blockIdx.x * blockDim.x + threadIdx.x;
    if (i >= N) return;
    float y = s4[i] * ndst[i] + b4[0];
    int g = graph_id[i];
    atomicAdd(gsum + g, y);
    atomicAdd(gcnt + g, 1.0f);
}

__global__ void finalize_kernel(const float* __restrict__ gsum, const float* __restrict__ gcnt,
                                float* __restrict__ out, int G) {
    int g = blockIdx.x * blockDim.x + threadIdx.x;
    if (g >= G) return;
    out[g] = gsum[g] / fmaxf(gcnt[g], 1.0f);
}

extern "C" void kernel_launch(void* const* d_in, const int* in_sizes, int n_in,
                              void* d_out, int out_size, void* d_ws, size_t ws_size,
                              hipStream_t stream) {
    const float* features = (const float*)d_in[0];
    const int*   src      = (const int*)d_in[1];
    const int*   dst      = (const int*)d_in[2];
    const int*   graph_id = (const int*)d_in[3];
    const float* W1 = (const float*)d_in[4];  const float* b1 = (const float*)d_in[5];
    const float* W2 = (const float*)d_in[6];  const float* b2 = (const float*)d_in[7];
    const float* W3 = (const float*)d_in[8];  const float* b3 = (const float*)d_in[9];
    const float* W4 = (const float*)d_in[10]; const float* b4 = (const float*)d_in[11];

    const int E = in_sizes[1];
    const int N = in_sizes[3];
    const int G = out_size;            // OUT_FEATS == 1
    float* out  = (float*)d_out;

    // workspace layout (floats)
    float* ws       = (float*)d_ws;
    float* norm_src = ws;                          // N  (deg_out -> rsqrt)
    float* norm_dst = ws + (size_t)N;              // N  (deg_in  -> rsqrt)
    float* agg      = ws + (size_t)2 * N;          // 128*N  (also hosts agg16 / s4 / t)
    float* act      = agg + (size_t)128 * N;       // 128*N  (also hosts gsum/gcnt at end)
    float* s4   = agg;                 // N floats, reused after layer-3 GEMM consumed agg
    float* tbuf = agg + (size_t)N;     // N floats
    float* gsum = act;                 // G floats, reused after dot128 consumed act
    float* gcnt = act + (size_t)G;     // G floats

    const int ZB = 2048;

    // degrees -> norms
    zero_f32<<<ZB, 256, 0, stream>>>(norm_src, (long long)2 * N);
    degree_kernel<<<(E + 255) / 256, 256, 0, stream>>>(src, dst, norm_src, norm_dst, E);
    norm_kernel<<<(N + 255) / 256, 256, 0, stream>>>(norm_src, norm_dst, N);

    // ---- layer 1: scatter at dim 16, then GEMM 16->128 (+ReLU) ----
    zero_f32<<<ZB, 256, 0, stream>>>(agg, (long long)16 * N);
    scatter16<<<((long long)E * 4 + 255) / 256, 256, 0, stream>>>(features, norm_src, src, dst, agg, E);
    gemm_norm_bias<16, 32, true><<<(N + 63) / 64, 128, 0, stream>>>(agg, norm_dst, W1, b1, act, N);

    // ---- layer 2: scatter at dim 128, GEMM 128->128 (+ReLU) ----
    zero_f32<<<ZB, 256, 0, stream>>>(agg, (long long)128 * N);
    scatter128<<<(E + 7) / 8, 256, 0, stream>>>(act, norm_src, src, dst, agg, E);
    gemm_norm_bias<128, 128, true><<<(N + 63) / 64, 128, 0, stream>>>(agg, norm_dst, W2, b2, act, N);

    // ---- layer 3 ----
    zero_f32<<<ZB, 256, 0, stream>>>(agg, (long long)128 * N);
    scatter128<<<(E + 7) / 8, 256, 0, stream>>>(act, norm_src, src, dst, agg, E);
    gemm_norm_bias<128, 128, true><<<(N + 63) / 64, 128, 0, stream>>>(agg, norm_dst, W3, b3, act, N);

    // ---- layer 4 (reordered): t = x @ W4 per node, then scalar scatter ----
    dot128<<<(N + 7) / 8, 256, 0, stream>>>(act, W4, tbuf, N);
    zero_f32<<<ZB, 256, 0, stream>>>(s4, (long long)N);
    scatter1<<<(E + 255) / 256, 256, 0, stream>>>(tbuf, norm_src, src, dst, s4, E);

    // ---- per-graph mean pooling ----
    zero_f32<<<(2 * G + 255) / 256, 256, 0, stream>>>(gsum, (long long)2 * G);
    pool_kernel<<<(N + 255) / 256, 256, 0, stream>>>(s4, norm_dst, graph_id, b4, gsum, gcnt, N);
    finalize_kernel<<<(G + 255) / 256, 256, 0, stream>>>(gsum, gcnt, out, G);
}
